// CounterFlowNetwork_74758200754574
// MI455X (gfx1250) — compile-verified, matrix operand order checked by reading, not run on hardware
//
#include <hip/hip_runtime.h>

// ---------------------------------------------------------------------------
// CounterFlowNetwork on MI455X (gfx1250): bf16 WMMA GEMMs, f32 accumulate.
// Each wave owns MT 16-row tiles (MT=2 for K=256 GEMMs) so every B-fragment
// loaded from L2 feeds MT WMMAs -> 2x arithmetic intensity vs. L2.
// LDS round-trips convert C-frag -> A-frag layout between chained GEMMs;
// sigmoid / alpha / residual epilogues are fused into the fragment stores.
// ---------------------------------------------------------------------------

typedef __attribute__((ext_vector_type(16))) __bf16 v16bf;
typedef __attribute__((ext_vector_type(8)))  float  v8f;

namespace {

constexpr int B_ROWS    = 16384;
constexpr int D_IN      = 512;
constexpr int D         = 256;    // d_gas == d_liquid
constexpr int D_OUT     = 1000;
constexpr int D_OUT_PAD = 1024;
constexpr int N_PLATES  = 8;
constexpr int N_SWEEPS  = 2;
constexpr int ROW_TILES = B_ROWS / 16;   // 1024 16-row tiles

constexpr int KS_D  = D    + 8;   // padded LDS row stride (bf16 elems), 16B aligned
constexpr int KS_IN = D_IN + 8;

__device__ __forceinline__ unsigned short f2bf(float f) {
    union { float f; unsigned u; } c; c.f = f;
    unsigned u = c.u;
    unsigned r = u + 0x7fffu + ((u >> 16) & 1u);   // round-to-nearest-even
    return (unsigned short)(r >> 16);
}
__device__ __forceinline__ float bf2f(unsigned short h) {
    union { unsigned u; float f; } c; c.u = ((unsigned)h) << 16;
    return c.f;
}

union AFrag { v16bf v; unsigned short s[16]; uint4 q[2]; };

// A-fragment (16x32 bf16) gather from an LDS tile with padded row stride `ks`.
// Per ISA layout: lane half h, element i -> K = kt*32 + (i<8 ? i : i+8) + 8*h.
__device__ __forceinline__ v16bf load_afrag_lds(const unsigned short* tile, int ks,
                                                int kt, int lane) {
    int m = lane & 15, h = lane >> 4;
    const unsigned short* base = tile + m * ks + kt * 32 + h * 8;
    AFrag u;
    u.q[0] = *(const uint4*)(base);        // K = 0..7  (+8h)
    u.q[1] = *(const uint4*)(base + 16);   // K = 16..23 (+8h)
    return u.v;
}

// B-fragment from a pre-swizzled weight array: [nt][kt][lane][16 bf16].
__device__ __forceinline__ v16bf load_bfrag(const unsigned short* wf, int nkt,
                                            int nt, int kt, int lane) {
    const uint4* p = (const uint4*)(wf + ((((size_t)nt * nkt + kt) * 32 + lane) << 4));
    AFrag u; u.q[0] = p[0]; u.q[1] = p[1];
    return u.v;
}

// Multi-row-tile GEMM: aTile holds MT*16 rows x K bf16 (padded stride ks);
// every B-fragment is reused across the MT row tiles (independent accs ->
// back-to-back v_wmma, no RAW hazard).
template <int NKT, int MT>
__device__ __forceinline__ void gemm_multi(const unsigned short* aTile, int ks,
                                           const unsigned short* wf, int nt, int lane,
                                           v8f acc[MT]) {
    // Cover next-nt B-fragment latency (lowers to global_prefetch_b8).
    __builtin_prefetch(wf + ((((size_t)(nt + 1) * NKT) * 32 + lane) << 4), 0, 0);
    #pragma unroll
    for (int kt = 0; kt < NKT; ++kt) {
        v16bf b = load_bfrag(wf, NKT, nt, kt, lane);
        #pragma unroll
        for (int mt = 0; mt < MT; ++mt) {
            v16bf a = load_afrag_lds(aTile + mt * 16 * ks, ks, kt, lane);
            acc[mt] = __builtin_amdgcn_wmma_f32_16x16x32_bf16(false, a, false, b,
                                                              (short)0, acc[mt], false, false);
        }
    }
}

// In-wave LDS ordering fence (CDNA5 split counters; DS ops are in-order per wave,
// the asm keeps the compiler from reordering across it).
__device__ __forceinline__ void lds_fence() {
    asm volatile("s_wait_dscnt 0" ::: "memory");
}

// Stage a ROWS x K bf16 tile (row-major, row stride K) from global into LDS.
template <int ROWS>
__device__ __forceinline__ void stage_bf16_tile(const unsigned short* src, size_t row0,
                                                int K, unsigned short* lds, int ks, int lane) {
    int cpr = K / 8;
    int total = ROWS * cpr;
    for (int c = lane; c < total; c += 32) {
        int r = c / cpr, cc = c - r * cpr;
        *(uint4*)(lds + r * ks + cc * 8) = *(const uint4*)(src + (row0 + r) * (size_t)K + cc * 8);
    }
}

template <int ROWS>
__device__ __forceinline__ void stage_zero_tile(unsigned short* lds, int K, int ks, int lane) {
    int cpr = K / 8;
    int total = ROWS * cpr;
    uint4 z = {0u, 0u, 0u, 0u};
    for (int c = lane; c < total; c += 32) {
        int r = c / cpr, cc = c - r * cpr;
        *(uint4*)(lds + r * ks + cc * 8) = z;
    }
}

template <int ROWS>
__device__ __forceinline__ void stage_f32_tile(const float* src, size_t row0, int K,
                                               unsigned short* lds, int ks, int lane) {
    int cpr = K / 4;
    int total = ROWS * cpr;
    for (int c = lane; c < total; c += 32) {
        int r = c / cpr, cc = c - r * cpr;
        float4 v = *(const float4*)(src + (row0 + r) * (size_t)K + cc * 4);
        unsigned short* d = lds + r * ks + cc * 4;
        d[0] = f2bf(v.x); d[1] = f2bf(v.y); d[2] = f2bf(v.z); d[3] = f2bf(v.w);
    }
}

// ---------------------------------------------------------------------------
// Weight swizzle: W (Kdim x NdimSrc f32, row-major) -> bf16 B-fragment layout
// [nt][kt][lane][i]; cols >= NdimSrc zero-padded (for W2 -> 1024).
// ---------------------------------------------------------------------------
__global__ void prep_weight_kernel(const float* __restrict__ W, unsigned short* __restrict__ out,
                                   int Kdim, int Ndim, int NdimSrc) {
    size_t idx = (size_t)blockIdx.x * blockDim.x + threadIdx.x;
    size_t total = (size_t)Kdim * Ndim;
    if (idx >= total) return;
    int i    = (int)(idx & 15);
    int lane = (int)((idx >> 4) & 31);
    size_t t = idx >> 9;                 // tile linear index = nt * nkt + kt
    int nkt = Kdim / 32;
    int kt = (int)(t % nkt);
    int nt = (int)(t / nkt);
    int n = nt * 16 + (lane & 15);
    int k = kt * 32 + (i < 8 ? i : i + 8) + 8 * (lane >> 4);
    float v = (n < NdimSrc) ? W[(size_t)k * NdimSrc + n] : 0.0f;
    out[idx] = f2bf(v);
}

// ---------------------------------------------------------------------------
// Encoder: g0 = relu(x @ W_ge + b_ge), x f32 [B,512] -> g0 bf16 [B,256]
// (K=512 -> one 16-row tile per wave; A-frags fill 128 VGPRs)
// ---------------------------------------------------------------------------
__global__ void __launch_bounds__(32)
encoder_kernel(const float* __restrict__ x, const unsigned short* __restrict__ Wge,
               const float* __restrict__ bge, unsigned short* __restrict__ g0) {
    __shared__ unsigned short xt[16 * KS_IN];
    int lane = threadIdx.x;
    size_t row0 = (size_t)blockIdx.x * 16;
    stage_f32_tile<16>(x, row0, D_IN, xt, KS_IN, lane);
    lds_fence();
    int nl = lane & 15, h = lane >> 4;
    for (int nt = 0; nt < D / 16; ++nt) {
        v8f acc[1] = {};
        gemm_multi<D_IN / 32, 1>(xt, KS_IN, Wge, nt, lane, acc);
        int n = nt * 16 + nl;
        float bias = bge[n];
        #pragma unroll
        for (int j = 0; j < 8; ++j) {
            float v = acc[0][j] + bias;
            v = v > 0.0f ? v : 0.0f;
            g0[(row0 + j + 8 * h) * D + n] = f2bf(v);
        }
    }
}

// ---------------------------------------------------------------------------
// Descending plate: l_out = l + alpha*((g - sigmoid(l@Weq+beq))@Wtr + btr)@Wab + bab
// 3 fused GEMMs, 32 rows (MT=2) per wave; C->A layout fix via LDS round-trips.
// ---------------------------------------------------------------------------
template <bool LZERO>
__global__ void __launch_bounds__(32)
plate_desc_kernel(const unsigned short* __restrict__ g_in,
                  const unsigned short* __restrict__ l_in,
                  unsigned short* __restrict__ l_out,
                  const unsigned short* __restrict__ Weq,
                  const unsigned short* __restrict__ Wtr,
                  const unsigned short* __restrict__ Wab,
                  const float* __restrict__ beq, const float* __restrict__ btr,
                  const float* __restrict__ bab, const float* __restrict__ alpha_p) {
    __shared__ unsigned short Lt[32 * KS_D];   // liquid tiles (kept for residual)
    __shared__ unsigned short Gt[32 * KS_D];   // gas tiles; reused to hold delta
    __shared__ unsigned short Ft[32 * KS_D];   // driving force df
    int lane = threadIdx.x;
    size_t row0 = (size_t)blockIdx.x * 32;
    if (LZERO) stage_zero_tile<32>(Lt, D, KS_D, lane);
    else       stage_bf16_tile<32>(l_in, row0, D, Lt, KS_D, lane);
    stage_bf16_tile<32>(g_in, row0, D, Gt, KS_D, lane);
    lds_fence();
    int nl = lane & 15, h = lane >> 4;
    float alpha = alpha_p[0];
    // GEMM1: e = sigmoid(l@Weq + beq); df = g - e
    for (int nt = 0; nt < D / 16; ++nt) {
        v8f acc[2] = {};
        gemm_multi<D / 32, 2>(Lt, KS_D, Weq, nt, lane, acc);
        int n = nt * 16 + nl;
        float bias = beq[n];
        #pragma unroll
        for (int mt = 0; mt < 2; ++mt) {
            #pragma unroll
            for (int j = 0; j < 8; ++j) {
                int m = mt * 16 + j + 8 * h;
                float e = 1.0f / (1.0f + __expf(-(acc[mt][j] + bias)));
                Ft[m * KS_D + n] = f2bf(bf2f(Gt[m * KS_D + n]) - e);
            }
        }
    }
    lds_fence();
    // GEMM2: delta = alpha*(df@Wtr + btr)  (g no longer needed -> overwrite Gt)
    for (int nt = 0; nt < D / 16; ++nt) {
        v8f acc[2] = {};
        gemm_multi<D / 32, 2>(Ft, KS_D, Wtr, nt, lane, acc);
        int n = nt * 16 + nl;
        float bias = btr[n];
        #pragma unroll
        for (int mt = 0; mt < 2; ++mt) {
            #pragma unroll
            for (int j = 0; j < 8; ++j) {
                int m = mt * 16 + j + 8 * h;
                Gt[m * KS_D + n] = f2bf(alpha * (acc[mt][j] + bias));
            }
        }
    }
    lds_fence();
    // GEMM3: l_out = l + delta@Wab + bab
    for (int nt = 0; nt < D / 16; ++nt) {
        v8f acc[2] = {};
        gemm_multi<D / 32, 2>(Gt, KS_D, Wab, nt, lane, acc);
        int n = nt * 16 + nl;
        float bias = bab[n];
        #pragma unroll
        for (int mt = 0; mt < 2; ++mt) {
            #pragma unroll
            for (int j = 0; j < 8; ++j) {
                int m = mt * 16 + j + 8 * h;
                float lv = bf2f(Lt[m * KS_D + n]);
                l_out[(row0 + m) * D + n] = f2bf(lv + acc[mt][j] + bias);
            }
        }
    }
}

// ---------------------------------------------------------------------------
// Ascending plate: g_out = g - alpha*((g - sigmoid(l@Weq+beq))@Wtr + btr)
// 2 fused GEMMs (Wab path not needed for gas update), 32 rows per wave.
// ---------------------------------------------------------------------------
__global__ void __launch_bounds__(32)
plate_asc_kernel(const unsigned short* __restrict__ g_in,
                 const unsigned short* __restrict__ l_in,
                 unsigned short* __restrict__ g_out,
                 const unsigned short* __restrict__ Weq,
                 const unsigned short* __restrict__ Wtr,
                 const float* __restrict__ beq, const float* __restrict__ btr,
                 const float* __restrict__ alpha_p) {
    __shared__ unsigned short Lt[32 * KS_D];
    __shared__ unsigned short Gt[32 * KS_D];
    __shared__ unsigned short Ft[32 * KS_D];
    int lane = threadIdx.x;
    size_t row0 = (size_t)blockIdx.x * 32;
    stage_bf16_tile<32>(l_in, row0, D, Lt, KS_D, lane);
    stage_bf16_tile<32>(g_in, row0, D, Gt, KS_D, lane);
    lds_fence();
    int nl = lane & 15, h = lane >> 4;
    float alpha = alpha_p[0];
    for (int nt = 0; nt < D / 16; ++nt) {
        v8f acc[2] = {};
        gemm_multi<D / 32, 2>(Lt, KS_D, Weq, nt, lane, acc);
        int n = nt * 16 + nl;
        float bias = beq[n];
        #pragma unroll
        for (int mt = 0; mt < 2; ++mt) {
            #pragma unroll
            for (int j = 0; j < 8; ++j) {
                int m = mt * 16 + j + 8 * h;
                float e = 1.0f / (1.0f + __expf(-(acc[mt][j] + bias)));
                Ft[m * KS_D + n] = f2bf(bf2f(Gt[m * KS_D + n]) - e);
            }
        }
    }
    lds_fence();
    for (int nt = 0; nt < D / 16; ++nt) {
        v8f acc[2] = {};
        gemm_multi<D / 32, 2>(Ft, KS_D, Wtr, nt, lane, acc);
        int n = nt * 16 + nl;
        float bias = btr[n];
        #pragma unroll
        for (int mt = 0; mt < 2; ++mt) {
            #pragma unroll
            for (int j = 0; j < 8; ++j) {
                int m = mt * 16 + j + 8 * h;
                float delta = alpha * (acc[mt][j] + bias);
                g_out[(row0 + m) * D + n] = f2bf(bf2f(Gt[m * KS_D + n]) - delta);
            }
        }
    }
}

// ---------------------------------------------------------------------------
// Head 1: h = relu(concat(g8, l1) @ W1 + b1)   (K = 512, MT=1)
// ---------------------------------------------------------------------------
__global__ void __launch_bounds__(32)
head1_kernel(const unsigned short* __restrict__ g8, const unsigned short* __restrict__ l1,
             const unsigned short* __restrict__ W1f, const float* __restrict__ b1,
             unsigned short* __restrict__ hbuf) {
    __shared__ unsigned short ft[16 * KS_IN];
    int lane = threadIdx.x;
    size_t row0 = (size_t)blockIdx.x * 16;
    stage_bf16_tile<16>(g8, row0, D, ft,       KS_IN, lane);   // cols   0..255
    stage_bf16_tile<16>(l1, row0, D, ft + 256, KS_IN, lane);   // cols 256..511
    lds_fence();
    int nl = lane & 15, h = lane >> 4;
    for (int nt = 0; nt < D / 16; ++nt) {
        v8f acc[1] = {};
        gemm_multi<D_IN / 32, 1>(ft, KS_IN, W1f, nt, lane, acc);
        int n = nt * 16 + nl;
        float bias = b1[n];
        #pragma unroll
        for (int j = 0; j < 8; ++j) {
            float v = acc[0][j] + bias;
            v = v > 0.0f ? v : 0.0f;
            hbuf[(row0 + j + 8 * h) * D + n] = f2bf(v);
        }
    }
}

// ---------------------------------------------------------------------------
// Head 2: out = h @ W2 + b2   (N = 1000 padded to 1024; MT=2; f32 output)
// ---------------------------------------------------------------------------
__global__ void __launch_bounds__(32)
head2_kernel(const unsigned short* __restrict__ hbuf, const unsigned short* __restrict__ W2f,
             const float* __restrict__ b2, float* __restrict__ out) {
    __shared__ unsigned short ht[32 * KS_D];
    int lane = threadIdx.x;
    size_t row0 = (size_t)blockIdx.x * 32;
    stage_bf16_tile<32>(hbuf, row0, D, ht, KS_D, lane);
    lds_fence();
    int nl = lane & 15, h = lane >> 4;
    for (int nt = 0; nt < D_OUT_PAD / 16; ++nt) {
        v8f acc[2] = {};
        gemm_multi<D / 32, 2>(ht, KS_D, W2f, nt, lane, acc);
        int n = nt * 16 + nl;
        if (n < D_OUT) {
            float bias = b2[n];
            #pragma unroll
            for (int mt = 0; mt < 2; ++mt) {
                #pragma unroll
                for (int j = 0; j < 8; ++j) {
                    out[(row0 + mt * 16 + j + 8 * h) * (size_t)D_OUT + n] = acc[mt][j] + bias;
                }
            }
        }
    }
}

} // anonymous namespace

// ---------------------------------------------------------------------------
// Host orchestration. Workspace (~146 MB): swizzled bf16 weights + 17 bf16
// activation state buffers (g[0..8], l[1..8]) + h. All launches on `stream`;
// fully deterministic, no host-side state.
// ---------------------------------------------------------------------------
extern "C" void kernel_launch(void* const* d_in, const int* in_sizes, int n_in,
                              void* d_out, int out_size, void* d_ws, size_t ws_size,
                              hipStream_t stream) {
    (void)in_sizes; (void)n_in; (void)out_size; (void)ws_size;
    const float* x     = (const float*)d_in[0];
    const float* W_ge  = (const float*)d_in[1];
    const float* b_ge  = (const float*)d_in[2];
    const float* W_eq  = (const float*)d_in[3];
    const float* b_eq  = (const float*)d_in[4];
    const float* W_tr  = (const float*)d_in[5];
    const float* b_tr  = (const float*)d_in[6];
    const float* W_ab  = (const float*)d_in[7];
    const float* b_ab  = (const float*)d_in[8];
    const float* alpha = (const float*)d_in[9];
    const float* W1    = (const float*)d_in[10];
    const float* b1    = (const float*)d_in[11];
    const float* W2    = (const float*)d_in[12];
    const float* b2    = (const float*)d_in[13];
    float* out = (float*)d_out;

    char* ws = (char*)d_ws;
    auto alloc = [&](size_t bytes) -> char* {
        char* p = ws; ws += (bytes + 255) & ~(size_t)255; return p;
    };
    const size_t ACT = (size_t)B_ROWS * D * sizeof(unsigned short);   // 8 MB
    unsigned short* Wge_f = (unsigned short*)alloc((size_t)D_IN * D * 2);
    unsigned short* Weq_f = (unsigned short*)alloc((size_t)D * D * 2);
    unsigned short* Wtr_f = (unsigned short*)alloc((size_t)D * D * 2);
    unsigned short* Wab_f = (unsigned short*)alloc((size_t)D * D * 2);
    unsigned short* W1_f  = (unsigned short*)alloc((size_t)(2 * D) * D * 2);
    unsigned short* W2_f  = (unsigned short*)alloc((size_t)D * D_OUT_PAD * 2);
    unsigned short* gbuf[N_PLATES + 1];
    for (int i = 0; i <= N_PLATES; ++i) gbuf[i] = (unsigned short*)alloc(ACT);
    unsigned short* lbuf[N_PLATES + 1];
    lbuf[0] = nullptr;
    for (int i = 1; i <= N_PLATES; ++i) lbuf[i] = (unsigned short*)alloc(ACT);
    unsigned short* hbuf = (unsigned short*)alloc(ACT);

    // --- weight swizzle to B-fragment bf16 layout ---
    {
        struct { const float* W; unsigned short* dst; int K, N, Nsrc; } jobs[6] = {
            { W_ge, Wge_f, D_IN,  D,         D     },
            { W_eq, Weq_f, D,     D,         D     },
            { W_tr, Wtr_f, D,     D,         D     },
            { W_ab, Wab_f, D,     D,         D     },
            { W1,   W1_f,  2 * D, D,         D     },
            { W2,   W2_f,  D,     D_OUT_PAD, D_OUT },
        };
        for (int j = 0; j < 6; ++j) {
            size_t total = (size_t)jobs[j].K * jobs[j].N;
            prep_weight_kernel<<<(unsigned)((total + 255) / 256), 256, 0, stream>>>(
                jobs[j].W, jobs[j].dst, jobs[j].K, jobs[j].N, jobs[j].Nsrc);
        }
    }

    // --- encoder: g0 ---
    encoder_kernel<<<ROW_TILES, 32, 0, stream>>>(x, Wge_f, b_ge, gbuf[0]);

    // --- counter-flow sweeps (MT=2 kernels use 32-row blocks) ---
    const int BLOCKS32 = ROW_TILES / 2;   // 512
    for (int s = 0; s < N_SWEEPS; ++s) {
        // descending liquid sweep: l[n] from (g[n-1] or g0, l[n+1]); l[9] == 0 always
        for (int n = N_PLATES; n >= 1; --n) {
            const unsigned short* gin = (s == 0) ? gbuf[0] : gbuf[n - 1];
            if (n == N_PLATES) {
                plate_desc_kernel<true><<<BLOCKS32, 32, 0, stream>>>(
                    gin, nullptr, lbuf[n], Weq_f, Wtr_f, Wab_f, b_eq, b_tr, b_ab, alpha);
            } else {
                plate_desc_kernel<false><<<BLOCKS32, 32, 0, stream>>>(
                    gin, lbuf[n + 1], lbuf[n], Weq_f, Wtr_f, Wab_f, b_eq, b_tr, b_ab, alpha);
            }
        }
        // ascending gas sweep: g[n] from (g[n-1], l[n])
        for (int n = 1; n <= N_PLATES; ++n) {
            plate_asc_kernel<<<BLOCKS32, 32, 0, stream>>>(
                gbuf[n - 1], lbuf[n], gbuf[n], Weq_f, Wtr_f, b_eq, b_tr, alpha);
        }
    }

    // --- output head ---
    head1_kernel<<<ROW_TILES, 32, 0, stream>>>(gbuf[N_PLATES], lbuf[1], W1_f, b1, hbuf);
    head2_kernel<<<BLOCKS32, 32, 0, stream>>>(hbuf, W2_f, b2, out);
}
